// FourierLayer3D_27977416966517
// MI455X (gfx1250) — matrix operand order repaired and checked by reading, not compiled, not run on hardware
//
#include <hip/hip_runtime.h>

// ---------------------------------------------------------------------------
// FNO-3D layer on gfx1250: partial-DFT pipeline, all heavy GEMMs on
// v_wmma_f32_16x16x32_bf16 (f32 accumulate, bf16 in-register conversion).
// Round 2: vectorized (b128/b64) fragment loads on the streaming kernels.
// ---------------------------------------------------------------------------

typedef __attribute__((ext_vector_type(16))) __bf16 v16bf;
typedef __attribute__((ext_vector_type(8)))  float  v8f;
typedef __attribute__((ext_vector_type(4)))  float  v4f;
typedef __attribute__((ext_vector_type(2)))  float  v2f;

#define PI2 6.283185307179586f

__device__ __forceinline__ v8f wmma_bf16(v16bf a, v16bf b, v8f c) {
  // (neg_a, A, neg_b, B, c_mod, C, reuse_a, reuse_b)
  return __builtin_amdgcn_wmma_f32_16x16x32_bf16(false, a, false, b, (short)0, c,
                                                 false, false);
}

__device__ __forceinline__ v8f zero8() {
  v8f z;
#pragma unroll
  for (int j = 0; j < 8; ++j) z[j] = 0.0f;
  return z;
}

__device__ __forceinline__ v4f ld4(const float* p) {
  return *reinterpret_cast<const v4f*>(p);
}

__device__ __forceinline__ v16bf cvt16(v4f q0, v4f q1, v4f q2, v4f q3) {
  v16bf a;
#pragma unroll
  for (int i = 0; i < 4; ++i) {
    a[i]      = (__bf16)q0[i];
    a[4 + i]  = (__bf16)q1[i];
    a[8 + i]  = (__bf16)q2[i];
    a[12 + i] = (__bf16)q3[i];
  }
  return a;
}

// A-fragment K index for element i (16-bit A 16x32 layout):
// lane<16: K = {0..7, 16..23};  lane>=16: K = {8..15, 24..31}
__device__ __forceinline__ int a_kmap(int lane, int i) {
  return ((lane & 16) >> 1) + (i & 7) + ((i & 8) << 1);
}

__device__ __forceinline__ int wave_id()  { return blockIdx.x * (blockDim.x >> 5) + (threadIdx.x >> 5); }
__device__ __forceinline__ int wave_cnt() { return gridDim.x * (blockDim.x >> 5); }

// ---------------------------------------------------------------------------
// 1) local path: out[b,o,s] = sum_i conv_w[o,i]*x[b,i,s] + bias[o]
//    GEMM M=32(o) x K=32(i) x N=s, bias via WMMA C operand.
// ---------------------------------------------------------------------------
__global__ void k_conv(const float* __restrict__ x, const float* __restrict__ cw,
                       const float* __restrict__ cb, float* __restrict__ out) {
  const int S = 262144;  // 64^3
  const int lane = threadIdx.x & 31;
  const int row0 = (lane >> 4) << 3, col = lane & 15;
  const int gw = wave_id(), nw = wave_cnt();

  v16bf a0, a1;  // conv_w rows o=0..15 / 16..31, cols i
#pragma unroll
  for (int i = 0; i < 16; ++i) {
    int k = a_kmap(lane, i);
    a0[i] = (__bf16)cw[(lane & 15) * 32 + k];
    a1[i] = (__bf16)cw[(16 + (lane & 15)) * 32 + k];
  }
  v8f cb0, cb1;
#pragma unroll
  for (int j = 0; j < 8; ++j) { cb0[j] = cb[row0 + j]; cb1[j] = cb[16 + row0 + j]; }

  for (int t = gw; t < 8 * 16384; t += nw) {
    int b = t >> 14, st = t & 16383;
    size_t s0 = (size_t)st * 16;
    const float* xb = x + (size_t)b * 32 * S;
    v16bf bf;  // B-frag: channel-strided rows, coalesced across lanes
#pragma unroll
    for (int i = 0; i < 16; ++i) {
      int k = a_kmap(lane, i);  // input channel
      bf[i] = (__bf16)xb[(size_t)k * S + s0 + col];
    }
    v8f c0 = wmma_bf16(a0, bf, cb0);
    v8f c1 = wmma_bf16(a1, bf, cb1);
    float* ob = out + (size_t)b * 32 * S + s0 + col;
#pragma unroll
    for (int j = 0; j < 8; ++j) {
      ob[(size_t)(row0 + j) * S] = c0[j];
      ob[(size_t)(16 + row0 + j) * S] = c1[j];
    }
  }
}

// ---------------------------------------------------------------------------
// 2) rFFT over W, keep kz<12:  XW[r, 2kz+p] ; r=(b,c,d,h), K=w(64), N=24
//    col 2kz = sum_w x*cos, col 2kz+1 = -sum_w x*sin
// ---------------------------------------------------------------------------
__global__ void k_fftw(const float* __restrict__ x, float* __restrict__ xw) {
  const int lane = threadIdx.x & 31;
  const int row0 = (lane >> 4) << 3, col = lane & 15;
  const int gw = wave_id(), nw = wave_cnt();

  v16bf bfr[2][2];  // [ntile][kstep], generated twiddles
  for (int nt = 0; nt < 2; ++nt)
    for (int st = 0; st < 2; ++st) {
#pragma unroll
      for (int i = 0; i < 16; ++i) {
        int n = nt * 16 + col;
        int w = st * 32 + a_kmap(lane, i);
        float v = 0.0f;
        if (n < 24) {
          int kz = n >> 1;
          float th = PI2 * (float)(kz * w) / 64.0f;
          v = (n & 1) ? -__sinf(th) : __cosf(th);
        }
        bfr[nt][st][i] = (__bf16)v;
      }
    }

  for (int rt = gw; rt < 65536; rt += nw) {
    // A rows are 64-float contiguous: b128 loads (16B aligned)
    const float* r = x + (size_t)rt * 1024 + (lane & 15) * 64 + ((lane & 16) >> 1);
    v16bf a0 = cvt16(ld4(r),      ld4(r + 4),  ld4(r + 16), ld4(r + 20));
    v16bf a1 = cvt16(ld4(r + 32), ld4(r + 36), ld4(r + 48), ld4(r + 52));
    v8f acc0 = zero8(), acc1 = zero8();
    acc0 = wmma_bf16(a0, bfr[0][0], acc0);
    acc0 = wmma_bf16(a1, bfr[0][1], acc0);
    acc1 = wmma_bf16(a0, bfr[1][0], acc1);
    acc1 = wmma_bf16(a1, bfr[1][1], acc1);
    float* orow = xw + (size_t)rt * 384;  // 16 rows x 24
#pragma unroll
    for (int j = 0; j < 8; ++j) {
      orow[(row0 + j) * 24 + col] = acc0[j];
      if (col < 8) orow[(row0 + j) * 24 + 16 + col] = acc1[j];
    }
  }
}

// ---------------------------------------------------------------------------
// 3) DFT over H (e^{-i}), keep ky<12. Slab=(b,c,d).
//    Re' = Sum_h Re*cos + Im*sin ; Im' = Sum_h Im*cos - Re*sin
//    XH layout: [b,c,d][reim(2)][ky(12)][kz(12)]
// ---------------------------------------------------------------------------
__global__ void k_ffth(const float* __restrict__ xw, float* __restrict__ xh) {
  const int lane = threadIdx.x & 31;
  const int row0 = (lane >> 4) << 3, col = lane & 15;
  const int gw = wave_id(), nw = wave_cnt();

  v16bf ac[2], as_[2], am[2];  // rows ky, cols h (2 K-steps)
  for (int st = 0; st < 2; ++st) {
#pragma unroll
    for (int i = 0; i < 16; ++i) {
      int h = st * 32 + a_kmap(lane, i);
      int ky = lane & 15;
      float th = PI2 * (float)(ky * h) / 64.0f;
      float c = __cosf(th), s = __sinf(th);
      ac[st][i] = (__bf16)c; as_[st][i] = (__bf16)s; am[st][i] = (__bf16)(-s);
    }
  }
  for (int t = gw; t < 16384; t += nw) {
    const float* base = xw + (size_t)t * 1536;  // 64 h x 24
    v16bf br[2][2];  // [reim][kstep]
    for (int st = 0; st < 2; ++st) {
#pragma unroll
      for (int i = 0; i < 16; ++i) {
        int h = st * 32 + a_kmap(lane, i);
        v2f p; p[0] = 0.0f; p[1] = 0.0f;
        if (col < 12) p = *reinterpret_cast<const v2f*>(base + h * 24 + 2 * col);
        br[0][st][i] = (__bf16)p[0]; br[1][st][i] = (__bf16)p[1];
      }
    }
    v8f are = zero8(), aim = zero8();
    for (int st = 0; st < 2; ++st) {
      are = wmma_bf16(ac[st], br[0][st], are);
      are = wmma_bf16(as_[st], br[1][st], are);
      aim = wmma_bf16(am[st], br[0][st], aim);
      aim = wmma_bf16(ac[st], br[1][st], aim);
    }
    float* ob = xh + (size_t)t * 288;
#pragma unroll
    for (int j = 0; j < 8; ++j) {
      int ky = row0 + j;
      if (ky < 12 && col < 12) {
        ob[ky * 12 + col] = are[j];
        ob[144 + ky * 12 + col] = aim[j];
      }
    }
  }
}

// ---------------------------------------------------------------------------
// 4) DFT over D (e^{-i}), keep kx<12. Slab=(b,c), 9 col-tiles over n=(ky,kz)=144.
//    XD layout: [b,c][reim(2)][kx(12)][n(144)]
// ---------------------------------------------------------------------------
__global__ void k_fftd(const float* __restrict__ xh, float* __restrict__ xd) {
  const int lane = threadIdx.x & 31;
  const int row0 = (lane >> 4) << 3, col = lane & 15;
  const int gw = wave_id(), nw = wave_cnt();

  v16bf ac[2], as_[2], am[2];  // rows kx, cols d
  for (int st = 0; st < 2; ++st) {
#pragma unroll
    for (int i = 0; i < 16; ++i) {
      int d = st * 32 + a_kmap(lane, i);
      int kx = lane & 15;
      float th = PI2 * (float)(kx * d) / 64.0f;
      float c = __cosf(th), s = __sinf(th);
      ac[st][i] = (__bf16)c; as_[st][i] = (__bf16)s; am[st][i] = (__bf16)(-s);
    }
  }
  for (int t = gw; t < 2304; t += nw) {
    int slab = t / 9, ct = t % 9;
    int n = ct * 16 + col;
    const float* base = xh + (size_t)slab * 18432;  // 64 d x 288
    v16bf br[2][2];
    for (int st = 0; st < 2; ++st) {
#pragma unroll
      for (int i = 0; i < 16; ++i) {
        int d = st * 32 + a_kmap(lane, i);
        float vr = 0.0f, vi = 0.0f;
        if (n < 144) { vr = base[d * 288 + n]; vi = base[d * 288 + 144 + n]; }
        br[0][st][i] = (__bf16)vr; br[1][st][i] = (__bf16)vi;
      }
    }
    v8f are = zero8(), aim = zero8();
    for (int st = 0; st < 2; ++st) {
      are = wmma_bf16(ac[st], br[0][st], are);
      are = wmma_bf16(as_[st], br[1][st], are);
      aim = wmma_bf16(am[st], br[0][st], aim);
      aim = wmma_bf16(ac[st], br[1][st], aim);
    }
    float* ob = xd + (size_t)slab * 3456;
#pragma unroll
    for (int j = 0; j < 8; ++j) {
      int kx = row0 + j;
      if (kx < 12 && n < 144) {
        ob[kx * 144 + n] = are[j];
        ob[1728 + kx * 144 + n] = aim[j];
      }
    }
  }
}

// ---------------------------------------------------------------------------
// 5) complex channel mix per mode (tiny: 0.11 GFLOP) -> VALU, coalesced over mode
// ---------------------------------------------------------------------------
__global__ void k_mix(const float* __restrict__ xd, const float* __restrict__ wts,
                      float* __restrict__ y) {
  const int lane = threadIdx.x & 31;
  const int gw = wave_id(), nw = wave_cnt();
  for (int t = gw; t < 13824; t += nw) {  // 256 (b,o) x 54 mode-chunks
    int bo = t / 54, mc = t % 54;
    int mode = mc * 32 + lane;            // < 1728
    int b = bo >> 5, o = bo & 31;
    float yr = 0.0f, yi = 0.0f;
#pragma unroll 4
    for (int i = 0; i < 32; ++i) {
      const float* xp = xd + (size_t)(b * 32 + i) * 3456;
      float xr = xp[mode], xi = xp[1728 + mode];
      v2f w = *reinterpret_cast<const v2f*>(wts + ((size_t)(i * 32 + o) * 1728 + mode) * 2);
      yr += xr * w[0] - xi * w[1];
      yi += xr * w[1] + xi * w[0];
    }
    float* yp = y + (size_t)(b * 32 + o) * 3456;
    yp[mode] = yr;
    yp[1728 + mode] = yi;
  }
}

// ---------------------------------------------------------------------------
// 6) inverse over D (e^{+i}): Re = Sum cos*Re - sin*Im ; Im = Sum sin*Re + cos*Im
//    Slab=(b,o), A rows d (4 mtiles), K=kx(12 pad 32). Z1: [b,o][d][reim][144]
// ---------------------------------------------------------------------------
__global__ void k_ifftd(const float* __restrict__ y, float* __restrict__ z1) {
  const int lane = threadIdx.x & 31;
  const int row0 = (lane >> 4) << 3, col = lane & 15;
  const int gw = wave_id(), nw = wave_cnt();
  const int mt = gw & 3;

  v16bf ac, as_, am;  // rows d = mt*16+(lane&15), K = kx
#pragma unroll
  for (int i = 0; i < 16; ++i) {
    int kx = a_kmap(lane, i);
    int d = mt * 16 + (lane & 15);
    float c = 0.0f, s = 0.0f;
    if (kx < 12) { float th = PI2 * (float)(kx * d) / 64.0f; c = __cosf(th); s = __sinf(th); }
    ac[i] = (__bf16)c; as_[i] = (__bf16)s; am[i] = (__bf16)(-s);
  }
  for (int t = gw >> 2; t < 2304; t += (nw >> 2)) {
    int slab = t / 9, ct = t % 9;
    int n = ct * 16 + col;
    const float* base = y + (size_t)slab * 3456;
    v16bf bre, bim;
#pragma unroll
    for (int i = 0; i < 16; ++i) {
      int kx = a_kmap(lane, i);
      float vr = 0.0f, vi = 0.0f;
      if (kx < 12 && n < 144) { vr = base[kx * 144 + n]; vi = base[1728 + kx * 144 + n]; }
      bre[i] = (__bf16)vr; bim[i] = (__bf16)vi;
    }
    v8f are = zero8(), aim = zero8();
    are = wmma_bf16(ac, bre, are);
    are = wmma_bf16(am, bim, are);
    aim = wmma_bf16(as_, bre, aim);
    aim = wmma_bf16(ac, bim, aim);
    float* ob = z1 + (size_t)slab * 18432;
#pragma unroll
    for (int j = 0; j < 8; ++j) {
      int d = mt * 16 + row0 + j;
      if (n < 144) {
        ob[d * 288 + n] = are[j];
        ob[d * 288 + 144 + n] = aim[j];
      }
    }
  }
}

// ---------------------------------------------------------------------------
// 7) inverse over H (e^{+i}). Slab=(b,o,d). Z2: [b,o,d][h][2kz+reim]
// ---------------------------------------------------------------------------
__global__ void k_iffth(const float* __restrict__ z1, float* __restrict__ z2) {
  const int lane = threadIdx.x & 31;
  const int row0 = (lane >> 4) << 3, col = lane & 15;
  const int gw = wave_id(), nw = wave_cnt();
  const int mt = gw & 3;

  v16bf ac, as_, am;  // rows h = mt*16+(lane&15), K = ky
#pragma unroll
  for (int i = 0; i < 16; ++i) {
    int ky = a_kmap(lane, i);
    int h = mt * 16 + (lane & 15);
    float c = 0.0f, s = 0.0f;
    if (ky < 12) { float th = PI2 * (float)(ky * h) / 64.0f; c = __cosf(th); s = __sinf(th); }
    ac[i] = (__bf16)c; as_[i] = (__bf16)s; am[i] = (__bf16)(-s);
  }
  for (int t = gw >> 2; t < 16384; t += (nw >> 2)) {
    const float* base = z1 + (size_t)t * 288;
    v16bf bre, bim;
#pragma unroll
    for (int i = 0; i < 16; ++i) {
      int ky = a_kmap(lane, i);
      float vr = 0.0f, vi = 0.0f;
      if (ky < 12 && col < 12) { vr = base[ky * 12 + col]; vi = base[144 + ky * 12 + col]; }
      bre[i] = (__bf16)vr; bim[i] = (__bf16)vi;
    }
    v8f are = zero8(), aim = zero8();
    are = wmma_bf16(ac, bre, are);
    are = wmma_bf16(am, bim, are);
    aim = wmma_bf16(as_, bre, aim);
    aim = wmma_bf16(ac, bim, aim);
    float* ob = z2 + (size_t)t * 1536;  // 64 h x 24
#pragma unroll
    for (int j = 0; j < 8; ++j) {
      int h = mt * 16 + row0 + j;
      if (col < 12) {
        ob[h * 24 + 2 * col] = are[j];
        ob[h * 24 + 2 * col + 1] = aim[j];
      }
    }
  }
}

// ---------------------------------------------------------------------------
// 8) inverse over W (irfft, alpha(0)=1, alpha(kz>0)=2, scale 1/64^3) fused with
//    the conv path already in d_out via the WMMA C accumulator.
// ---------------------------------------------------------------------------
__global__ void k_ifftw(const float* __restrict__ z2, float* __restrict__ out) {
  const int lane = threadIdx.x & 31;
  const int row0 = (lane >> 4) << 3, col = lane & 15;
  const int gw = wave_id(), nw = wave_cnt();
  const float s3 = 1.0f / 262144.0f;  // 1/64^3
  const int hi = (lane >> 4) & 1;

  v16bf bfr[4];  // [ntile of w], K = 2kz+p (24 pad 32)
  for (int nt = 0; nt < 4; ++nt) {
#pragma unroll
    for (int i = 0; i < 16; ++i) {
      int w = nt * 16 + col;
      int k = a_kmap(lane, i);
      float v = 0.0f;
      if (k < 24) {
        int kz = k >> 1;
        float al = (kz ? 2.0f : 1.0f) * s3;
        float th = PI2 * (float)(kz * w) / 64.0f;
        v = (k & 1) ? -al * __sinf(th) : al * __cosf(th);
      }
      bfr[nt][i] = (__bf16)v;
    }
  }
  for (int rt = gw; rt < 65536; rt += nw) {
    // A rows are 24-float; lane<16 reads k {0..7,16..23}, lane>=16 reads
    // k {8..15} + zero-pad (k 24..31). All b128-aligned (96B row stride).
    const float* r0 = z2 + (size_t)rt * 384 + (lane & 15) * 24;
    v4f q0 = ld4(r0 + hi * 8), q1 = ld4(r0 + hi * 8 + 4);
    v4f q2 = ld4(r0 + 16),     q3 = ld4(r0 + 20);
    v16bf a;
#pragma unroll
    for (int i = 0; i < 4; ++i) {
      a[i]      = (__bf16)q0[i];
      a[4 + i]  = (__bf16)q1[i];
      a[8 + i]  = (__bf16)(hi ? 0.0f : q2[i]);
      a[12 + i] = (__bf16)(hi ? 0.0f : q3[i]);
    }
    float* ob = out + (size_t)rt * 1024;
    for (int nt = 0; nt < 4; ++nt) {
      v8f c;
#pragma unroll
      for (int j = 0; j < 8; ++j) c[j] = ob[(row0 + j) * 64 + nt * 16 + col];
      c = wmma_bf16(a, bfr[nt], c);
#pragma unroll
      for (int j = 0; j < 8; ++j) ob[(row0 + j) * 64 + nt * 16 + col] = c[j];
    }
  }
}

// ---------------------------------------------------------------------------
extern "C" void kernel_launch(void* const* d_in, const int* in_sizes, int n_in,
                              void* d_out, int out_size, void* d_ws, size_t ws_size,
                              hipStream_t stream) {
  (void)in_sizes; (void)n_in; (void)out_size; (void)ws_size;
  const float* x  = (const float*)d_in[0];
  const float* wt = (const float*)d_in[1];
  const float* cw = (const float*)d_in[2];
  const float* cb = (const float*)d_in[3];
  float* out = (float*)d_out;
  float* ws  = (float*)d_ws;

  // workspace layout (floats); Z1 aliases XH, Z2 aliases XW (stream-ordered safe)
  float* XW = ws;              // 25,165,824  [b,c,d,h][24]
  float* XH = ws + 25165824;   //  4,718,592  [b,c,d][2][12][12]
  float* XD = ws + 29884416;   //    884,736  [b,c][2][12][144]
  float* Y  = ws + 30769152;   //    884,736  [b,o][2][12][144]
  float* Z1 = XH;              // [b,o][d][2][144]
  float* Z2 = XW;              // [b,o,d][h][24]

  dim3 blk(256);
  k_conv <<<1024, blk, 0, stream>>>(x, cw, cb, out);
  k_fftw <<<1024, blk, 0, stream>>>(x, XW);
  k_ffth <<<1024, blk, 0, stream>>>(XW, XH);
  k_fftd <<<288,  blk, 0, stream>>>(XH, XD);
  k_mix  <<<1728, blk, 0, stream>>>(XD, wt, Y);
  k_ifftd<<<1152, blk, 0, stream>>>(Y, Z1);
  k_iffth<<<1024, blk, 0, stream>>>(Z1, Z2);
  k_ifftw<<<1024, blk, 0, stream>>>(Z2, out);
}
